// TransformerEncoderLayer_32066225832369
// MI455X (gfx1250) — matrix operand order repaired and checked by reading, not compiled, not run on hardware
//
#include <hip/hip_runtime.h>
#include <hip/hip_bf16.h>

// ---------------------------------------------------------------------------
// Problem constants (from reference): B=2, S=2048, D=768, H=12, DK=64, DFF=3072
// ---------------------------------------------------------------------------
static constexpr int kB   = 2;
static constexpr int kS   = 2048;
static constexpr int kD   = 768;
static constexpr int kH   = 12;
static constexpr int kDK  = 64;
static constexpr int kDFF = 3072;
static constexpr int kBS  = kB * kS;          // 4096 rows
static constexpr float kEPS = 1e-5f;

typedef __bf16 bf16_t;
typedef __attribute__((ext_vector_type(8)))  bf16_t v8bf;
typedef __attribute__((ext_vector_type(16))) bf16_t v16bf;
typedef __attribute__((ext_vector_type(8)))  float  v8f;

__device__ __forceinline__ v8f vzero8() {
    v8f z;
#pragma unroll
    for (int i = 0; i < 8; ++i) z[i] = 0.0f;
    return z;
}

// A-operand (16x32 bf16) per-lane load: halves 0..7 at K=kb..kb+7,
// halves 8..15 at K=kb+16..kb+23  (kb = (lane>>4)*8), per ISA 7.12.2.
__device__ __forceinline__ v16bf ld_a(const bf16_t* p0, const bf16_t* p1) {
    v8bf lo = *(const v8bf*)p0;
    v8bf hi = *(const v8bf*)p1;
    v16bf r;
#pragma unroll
    for (int i = 0; i < 8; ++i) { r[i] = lo[i]; r[8 + i] = hi[i]; }
    return r;
}

// B-operand (32x16 bf16) per-lane load: halves 0..7 at K=kb..kb+7,
// halves 8..15 at K=kb+16..kb+23  (kb = (lane>>4)*8, col = lane&15).
__device__ __forceinline__ v16bf ld_b(const bf16_t* p) {
    v8bf lo = *(const v8bf*)p;
    v8bf hi = *(const v8bf*)(p + 16);
    v16bf r;
#pragma unroll
    for (int i = 0; i < 8; ++i) { r[i] = lo[i]; r[8 + i] = hi[i]; }
    return r;
}

#define WMMA_BF16(a, b, c) \
    __builtin_amdgcn_wmma_f32_16x16x32_bf16(false, (a), false, (b), (short)0, (c), false, false)

// ---------------------------------------------------------------------------
// fp32 -> bf16 elementwise convert
// ---------------------------------------------------------------------------
__global__ void f2bf_kernel(const float* __restrict__ in, bf16_t* __restrict__ out, int n) {
    int i = blockIdx.x * 256 + threadIdx.x;
    if (i < n) out[i] = (bf16_t)in[i];
}

// Transpose [rows, cols] fp32 -> bf16 [cols, rows], batched over blockIdx.y.
__global__ void transpose_f2bf_kernel(const float* __restrict__ in, bf16_t* __restrict__ out,
                                      int rows, int cols) {
    size_t base = (size_t)blockIdx.y * rows * cols;
    int idx = blockIdx.x * 256 + threadIdx.x;
    if (idx < rows * cols) {
        int r = idx / cols, c = idx % cols;
        out[base + (size_t)c * rows + r] = (bf16_t)in[base + idx];
    }
}

// ---------------------------------------------------------------------------
// QKV projection: X[4096,768] @ Wt[h][64,768]^T -> per-head Q/K/V (bf16)
// grid = (BS/128, H, 3), block = 128 (4 waves, each wave: 32 rows x 64 cols,
// double-buffered K loop, 16 WMMA per 32-K step pair).
// which==2 (V) is stored transposed: Vt[(bh*DK + dk)*S + s]
// ---------------------------------------------------------------------------
__global__ __launch_bounds__(128)
void qkv_kernel(const bf16_t* __restrict__ Xb,
                const bf16_t* __restrict__ Wqt, const bf16_t* __restrict__ Wkt,
                const bf16_t* __restrict__ Wvt,
                const float* __restrict__ bq, const float* __restrict__ bk,
                const float* __restrict__ bv,
                bf16_t* __restrict__ Qb, bf16_t* __restrict__ Kb, bf16_t* __restrict__ Vtb) {
    const int wave = threadIdx.x >> 5;
    const int lane = threadIdx.x & 31;
    const int h = blockIdx.y;
    const int which = blockIdx.z;
    const int m0 = blockIdx.x * 128 + wave * 32;
    const int row = lane & 15, grp = lane >> 4;

    const bf16_t* Wt = (which == 0) ? Wqt : (which == 1) ? Wkt : Wvt;
    const float* bias = ((which == 0) ? bq : (which == 1) ? bk : bv) + h * kDK;
    Wt += (size_t)h * kDK * kD;

    const bf16_t* ar0 = Xb + (size_t)(m0 + row) * kD + grp * 8;
    const bf16_t* ar1 = ar0 + (size_t)16 * kD;
    const bf16_t* bp0 = Wt + (size_t)row * kD + grp * 8;

    v8f acc[2][4];
#pragma unroll
    for (int i = 0; i < 2; ++i)
#pragma unroll
        for (int t = 0; t < 4; ++t) acc[i][t] = vzero8();

    v16bf a0[2], b0[4], a1[2], b1[4];
    a0[0] = ld_a(ar0, ar0 + 16);
    a0[1] = ld_a(ar1, ar1 + 16);
#pragma unroll
    for (int t = 0; t < 4; ++t) b0[t] = ld_b(bp0 + (size_t)(t * 16) * kD);

    for (int k0 = 0; k0 < kD; k0 += 64) {
        const int ka = k0 + 32;
        a1[0] = ld_a(ar0 + ka, ar0 + ka + 16);
        a1[1] = ld_a(ar1 + ka, ar1 + ka + 16);
#pragma unroll
        for (int t = 0; t < 4; ++t) b1[t] = ld_b(bp0 + (size_t)(t * 16) * kD + ka);
#pragma unroll
        for (int i = 0; i < 2; ++i)
#pragma unroll
            for (int t = 0; t < 4; ++t) acc[i][t] = WMMA_BF16(a0[i], b0[t], acc[i][t]);

        const int kn = (k0 + 64 < kD) ? k0 + 64 : 0;   // clamp: dead prefetch on last iter
        a0[0] = ld_a(ar0 + kn, ar0 + kn + 16);
        a0[1] = ld_a(ar1 + kn, ar1 + kn + 16);
#pragma unroll
        for (int t = 0; t < 4; ++t) b0[t] = ld_b(bp0 + (size_t)(t * 16) * kD + kn);
#pragma unroll
        for (int i = 0; i < 2; ++i)
#pragma unroll
            for (int t = 0; t < 4; ++t) acc[i][t] = WMMA_BF16(a1[i], b1[t], acc[i][t]);
    }

#pragma unroll
    for (int i = 0; i < 2; ++i) {
#pragma unroll
        for (int t = 0; t < 4; ++t) {
            int col = t * 16 + (lane & 15);          // dk in [0,64)
            float bb = bias[col];
#pragma unroll
            for (int r = 0; r < 8; ++r) {
                int g = m0 + i * 16 + r + grp * 8;   // global row in [0, B*S)
                int bidx = g >> 11;                  // / S
                int s = g & (kS - 1);
                float v = acc[i][t][r] + bb;
                size_t bh = (size_t)bidx * kH + h;
                if (which == 2) {
                    Vtb[(bh * kDK + col) * kS + s] = (bf16_t)v;
                } else {
                    bf16_t* O = which ? Kb : Qb;
                    O[(bh * kS + s) * kDK + col] = (bf16_t)v;
                }
            }
        }
    }
}

// ---------------------------------------------------------------------------
// Flash attention: per wave: 16 query rows, online softmax over 32-key steps.
// V tiles are loaded before the softmax so the exp/shuffle chain hides their
// latency. grid = (S/64, B*H), block = 128 (4 waves)
// Output: head-concatenated bf16 Ctx[(b*S+s)*768 + h*64 + dk]
// ---------------------------------------------------------------------------
__global__ __launch_bounds__(128)
void attn_kernel(const bf16_t* __restrict__ Qb, const bf16_t* __restrict__ Kb,
                 const bf16_t* __restrict__ Vtb, bf16_t* __restrict__ Ctx) {
    __shared__ float pbuf[4][16 * 32];           // per-wave P bounce buffer
    const int wave = threadIdx.x >> 5;
    const int lane = threadIdx.x & 31;
    const int bh = blockIdx.y;                   // b*H + h
    const int m0 = blockIdx.x * 64 + wave * 16;  // query row in sequence
    const int row = lane & 15, grp = lane >> 4;
    const float scale = 0.125f;                  // 1/sqrt(64)

    const bf16_t* qrow = Qb + ((size_t)bh * kS + m0 + row) * kDK + grp * 8;
    const v16bf qa0 = ld_a(qrow, qrow + 16);          // K 0..31
    const v16bf qa1 = ld_a(qrow + 32, qrow + 48);     // K 32..63

    float mstate[8], lstate[8];
    v8f ctx[4];
#pragma unroll
    for (int r = 0; r < 8; ++r) { mstate[r] = -1e30f; lstate[r] = 0.0f; }
#pragma unroll
    for (int d = 0; d < 4; ++d) ctx[d] = vzero8();

    float* myp = pbuf[wave];

    for (int j = 0; j < kS; j += 32) {
        // ---- scores for keys [j, j+32) : two 16x16 tiles, 2 WMMA each ----
        v8f s0 = vzero8(), s1 = vzero8();
        {
            const bf16_t* kp = Kb + ((size_t)bh * kS + j + row) * kDK + grp * 8;
            v16bf kb0 = ld_b(kp);
            v16bf kb1 = ld_b(kp + 32);
            s0 = WMMA_BF16(qa0, kb0, s0);
            s0 = WMMA_BF16(qa1, kb1, s0);
            kp += 16 * kDK;
            kb0 = ld_b(kp);
            kb1 = ld_b(kp + 32);
            s1 = WMMA_BF16(qa0, kb0, s1);
            s1 = WMMA_BF16(qa1, kb1, s1);
        }
        // ---- issue V loads now; softmax VALU work below hides their latency
        v16bf bv[4];
#pragma unroll
        for (int d = 0; d < 4; ++d) {
            const bf16_t* vp =
                Vtb + ((size_t)bh * kDK + d * 16 + (lane & 15)) * kS + j + grp * 8;
            bv[d] = ld_b(vp);
        }
        // ---- online softmax (row = r + grp*8 lives in a 16-lane group) ----
        float alpha[8];
#pragma unroll
        for (int r = 0; r < 8; ++r) {
            float a0 = s0[r] * scale;
            float a1 = s1[r] * scale;
            float mt = fmaxf(a0, a1);
            mt = fmaxf(mt, __shfl_xor(mt, 1, 16));
            mt = fmaxf(mt, __shfl_xor(mt, 2, 16));
            mt = fmaxf(mt, __shfl_xor(mt, 4, 16));
            mt = fmaxf(mt, __shfl_xor(mt, 8, 16));
            float mnew = fmaxf(mstate[r], mt);
            float al = __expf(mstate[r] - mnew);
            float p0 = __expf(a0 - mnew);
            float p1 = __expf(a1 - mnew);
            float rs = p0 + p1;
            rs += __shfl_xor(rs, 1, 16);
            rs += __shfl_xor(rs, 2, 16);
            rs += __shfl_xor(rs, 4, 16);
            rs += __shfl_xor(rs, 8, 16);
            lstate[r] = lstate[r] * al + rs;
            mstate[r] = mnew;
            alpha[r] = al;
            // store P (C layout -> LDS [qrow][key])
            myp[(r + grp * 8) * 32 + (lane & 15)]      = p0;
            myp[(r + grp * 8) * 32 + 16 + (lane & 15)] = p1;
        }
        // rescale accumulated context
#pragma unroll
        for (int d = 0; d < 4; ++d)
#pragma unroll
            for (int r = 0; r < 8; ++r) ctx[d][r] = ctx[d][r] * alpha[r];

        // ---- read P back in A-operand layout (LDS is in-order per wave) ----
        v16bf pa;
        {
            const int ar = lane & 15;
            const int kb = grp * 8;
#pragma unroll
            for (int i = 0; i < 8; ++i) pa[i] = (bf16_t)myp[ar * 32 + kb + i];
#pragma unroll
            for (int i = 0; i < 8; ++i) pa[8 + i] = (bf16_t)myp[ar * 32 + 16 + kb + i];
        }
        // ---- ctx += P(16x32) @ V(32x64): 4 WMMA ----
#pragma unroll
        for (int d = 0; d < 4; ++d) ctx[d] = WMMA_BF16(pa, bv[d], ctx[d]);
    }

    // ---- normalize and store head-concatenated ----
    const int b = bh / kH, h = bh % kH;
#pragma unroll
    for (int d = 0; d < 4; ++d) {
#pragma unroll
        for (int r = 0; r < 8; ++r) {
            int srow = m0 + r + grp * 8;
            float v = ctx[d][r] / lstate[r];
            Ctx[((size_t)b * kS + srow) * (kH * kDK) + h * kDK + d * 16 + (lane & 15)] =
                (bf16_t)v;
        }
    }
}

// ---------------------------------------------------------------------------
// Generic bf16 WMMA GEMM: C[M,N] = A[M,K] @ Bt[N,K]^T (+bias, relu, residual)
// block = 256 (8 waves): 4 row-waves x 2 col-waves.
// Wave tile = 32x64 (8 WMMA / 32-K step), register double buffering over K
// (unroll by 64). Block tile = 128x128. grid = (M/128, N/128). K % 64 == 0.
// ---------------------------------------------------------------------------
__global__ __launch_bounds__(256)
void gemm_bf16_kernel(const bf16_t* __restrict__ A, const bf16_t* __restrict__ Bt,
                      const float* __restrict__ bias, const float* __restrict__ resid,
                      float* __restrict__ outF, bf16_t* __restrict__ outB,
                      int M, int N, int K, int relu) {
    const int wave = threadIdx.x >> 5;
    const int lane = threadIdx.x & 31;
    const int rw = wave & 3, cw = wave >> 2;
    const int m0 = blockIdx.x * 128 + rw * 32;
    const int n0 = blockIdx.y * 128 + cw * 64;
    const int row = lane & 15, grp = lane >> 4;

    const bf16_t* ar0 = A + (size_t)(m0 + row) * K + grp * 8;
    const bf16_t* ar1 = ar0 + (size_t)16 * K;
    const bf16_t* bp0 = Bt + (size_t)(n0 + row) * K + grp * 8;

    v8f acc[2][4];
#pragma unroll
    for (int i = 0; i < 2; ++i)
#pragma unroll
        for (int t = 0; t < 4; ++t) acc[i][t] = vzero8();

    v16bf a0[2], b0[4], a1[2], b1[4];
    a0[0] = ld_a(ar0, ar0 + 16);
    a0[1] = ld_a(ar1, ar1 + 16);
#pragma unroll
    for (int t = 0; t < 4; ++t) b0[t] = ld_b(bp0 + (size_t)(t * 16) * K);

    for (int k0 = 0; k0 < K; k0 += 64) {
        const int ka = k0 + 32;
        a1[0] = ld_a(ar0 + ka, ar0 + ka + 16);
        a1[1] = ld_a(ar1 + ka, ar1 + ka + 16);
#pragma unroll
        for (int t = 0; t < 4; ++t) b1[t] = ld_b(bp0 + (size_t)(t * 16) * K + ka);
#pragma unroll
        for (int i = 0; i < 2; ++i)
#pragma unroll
            for (int t = 0; t < 4; ++t) acc[i][t] = WMMA_BF16(a0[i], b0[t], acc[i][t]);

        const int kn = (k0 + 64 < K) ? k0 + 64 : 0;    // clamp: dead prefetch on last iter
        a0[0] = ld_a(ar0 + kn, ar0 + kn + 16);
        a0[1] = ld_a(ar1 + kn, ar1 + kn + 16);
#pragma unroll
        for (int t = 0; t < 4; ++t) b0[t] = ld_b(bp0 + (size_t)(t * 16) * K + kn);
#pragma unroll
        for (int i = 0; i < 2; ++i)
#pragma unroll
            for (int t = 0; t < 4; ++t) acc[i][t] = WMMA_BF16(a1[i], b1[t], acc[i][t]);
    }

#pragma unroll
    for (int i = 0; i < 2; ++i) {
#pragma unroll
        for (int t = 0; t < 4; ++t) {
            int ccol = n0 + t * 16 + (lane & 15);
            float bb = bias ? bias[ccol] : 0.0f;
#pragma unroll
            for (int r = 0; r < 8; ++r) {
                int crow = m0 + i * 16 + r + grp * 8;
                float v = acc[i][t][r] + bb;
                if (relu) v = fmaxf(v, 0.0f);
                if (resid) v += resid[(size_t)crow * N + ccol];
                if (outF) outF[(size_t)crow * N + ccol] = v;
                if (outB) outB[(size_t)crow * N + ccol] = (bf16_t)v;
            }
        }
    }
}

// ---------------------------------------------------------------------------
// LayerNorm over D=768 per row. One block (256 threads) per row.
// ---------------------------------------------------------------------------
__global__ __launch_bounds__(256)
void ln_kernel(const float* __restrict__ in, const float* __restrict__ w,
               const float* __restrict__ b, float* __restrict__ outF,
               bf16_t* __restrict__ outB, int N) {
    __shared__ float p1[8], p2[8];
    const int wave = threadIdx.x >> 5;
    const int lane = threadIdx.x & 31;
    const float* x = in + (size_t)blockIdx.x * N;

    float s = 0.0f, s2 = 0.0f;
    for (int i = threadIdx.x; i < N; i += 256) {
        float v = x[i];
        s += v;
        s2 += v * v;
    }
#pragma unroll
    for (int m = 16; m >= 1; m >>= 1) {
        s  += __shfl_xor(s, m, 32);
        s2 += __shfl_xor(s2, m, 32);
    }
    if (lane == 0) { p1[wave] = s; p2[wave] = s2; }
    __syncthreads();
    float tot = 0.0f, tot2 = 0.0f;
#pragma unroll
    for (int i = 0; i < 8; ++i) { tot += p1[i]; tot2 += p2[i]; }
    float mu = tot / N;
    float var = tot2 / N - mu * mu;
    float rstd = rsqrtf(var + kEPS);

    for (int i = threadIdx.x; i < N; i += 256) {
        float v = (x[i] - mu) * rstd * w[i] + b[i];
        if (outF) outF[(size_t)blockIdx.x * N + i] = v;
        if (outB) outB[(size_t)blockIdx.x * N + i] = (bf16_t)v;
    }
}

// ---------------------------------------------------------------------------
// Host orchestration
// ---------------------------------------------------------------------------
extern "C" void kernel_launch(void* const* d_in, const int* in_sizes, int n_in,
                              void* d_out, int out_size, void* d_ws, size_t ws_size,
                              hipStream_t stream) {
    const float* src  = (const float*)d_in[0];
    const float* Wq   = (const float*)d_in[1];
    const float* bq   = (const float*)d_in[2];
    const float* Wk   = (const float*)d_in[3];
    const float* bk   = (const float*)d_in[4];
    const float* Wv   = (const float*)d_in[5];
    const float* bv   = (const float*)d_in[6];
    const float* Wo   = (const float*)d_in[7];
    const float* bo   = (const float*)d_in[8];
    const float* ln1w = (const float*)d_in[9];
    const float* ln1b = (const float*)d_in[10];
    const float* W1   = (const float*)d_in[11];
    const float* b1   = (const float*)d_in[12];
    const float* W2   = (const float*)d_in[13];
    const float* b2   = (const float*)d_in[14];
    const float* ln2w = (const float*)d_in[15];
    const float* ln2b = (const float*)d_in[16];

    char* ws = (char*)d_ws;
    size_t off = 0;
    auto alloc = [&](size_t bytes) -> void* {
        void* p = ws + off;
        off += (bytes + 255) & ~(size_t)255;
        return p;
    };

    bf16_t* Xb   = (bf16_t*)alloc((size_t)kBS * kD * 2);
    bf16_t* Wqt  = (bf16_t*)alloc((size_t)kH * kD * kDK * 2);
    bf16_t* Wkt  = (bf16_t*)alloc((size_t)kH * kD * kDK * 2);
    bf16_t* Wvt  = (bf16_t*)alloc((size_t)kH * kD * kDK * 2);
    bf16_t* Wot  = (bf16_t*)alloc((size_t)kD * kD * 2);
    bf16_t* W1t  = (bf16_t*)alloc((size_t)kD * kDFF * 2);
    bf16_t* W2t  = (bf16_t*)alloc((size_t)kDFF * kD * 2);
    bf16_t* Qb   = (bf16_t*)alloc((size_t)kB * kH * kS * kDK * 2);
    bf16_t* Kb   = (bf16_t*)alloc((size_t)kB * kH * kS * kDK * 2);
    bf16_t* Vtb  = (bf16_t*)alloc((size_t)kB * kH * kS * kDK * 2);
    bf16_t* Ctx  = (bf16_t*)alloc((size_t)kBS * kD * 2);
    float*  R1   = (float*)alloc((size_t)kBS * kD * 4);
    float*  X1f  = (float*)alloc((size_t)kBS * kD * 4);
    bf16_t* X1b  = (bf16_t*)alloc((size_t)kBS * kD * 2);
    bf16_t* H1   = (bf16_t*)alloc((size_t)kBS * kDFF * 2);
    float*  R2   = (float*)alloc((size_t)kBS * kD * 4);
    (void)ws_size; (void)n_in; (void)in_sizes; (void)out_size;

    // 1) src -> bf16
    {
        int n = kBS * kD;
        f2bf_kernel<<<dim3((n + 255) / 256), dim3(256), 0, stream>>>(src, Xb, n);
    }
    // 2) weight transposes (fp32 -> bf16, [rows,cols] -> [cols,rows])
    {
        int n = kD * kDK;
        dim3 g((n + 255) / 256, kH);
        transpose_f2bf_kernel<<<g, 256, 0, stream>>>(Wq, Wqt, kD, kDK);
        transpose_f2bf_kernel<<<g, 256, 0, stream>>>(Wk, Wkt, kD, kDK);
        transpose_f2bf_kernel<<<g, 256, 0, stream>>>(Wv, Wvt, kD, kDK);
    }
    transpose_f2bf_kernel<<<dim3((kD * kD + 255) / 256, 1), 256, 0, stream>>>(Wo, Wot, kD, kD);
    transpose_f2bf_kernel<<<dim3((kD * kDFF + 255) / 256, 1), 256, 0, stream>>>(W1, W1t, kD, kDFF);
    transpose_f2bf_kernel<<<dim3((kDFF * kD + 255) / 256, 1), 256, 0, stream>>>(W2, W2t, kDFF, kD);

    // 3) QKV projections
    qkv_kernel<<<dim3(kBS / 128, kH, 3), dim3(128), 0, stream>>>(
        Xb, Wqt, Wkt, Wvt, bq, bk, bv, Qb, Kb, Vtb);

    // 4) attention
    attn_kernel<<<dim3(kS / 64, kB * kH), dim3(128), 0, stream>>>(Qb, Kb, Vtb, Ctx);

    // 5) output projection + bias + residual(src) -> R1 (fp32)
    gemm_bf16_kernel<<<dim3(kBS / 128, kD / 128), dim3(256), 0, stream>>>(
        Ctx, Wot, bo, src, R1, (bf16_t*)nullptr, kBS, kD, kD, 0);

    // 6) LN1 -> X1f (fp32) + X1b (bf16)
    ln_kernel<<<dim3(kBS), dim3(256), 0, stream>>>(R1, ln1w, ln1b, X1f, X1b, kD);

    // 7) FFN1: relu(X1 @ W1 + b1) -> H1 (bf16)
    gemm_bf16_kernel<<<dim3(kBS / 128, kDFF / 128), dim3(256), 0, stream>>>(
        X1b, W1t, b1, (const float*)nullptr, (float*)nullptr, H1, kBS, kDFF, kD, 1);

    // 8) FFN2: H1 @ W2 + b2 + X1f -> R2 (fp32)
    gemm_bf16_kernel<<<dim3(kBS / 128, kD / 128), dim3(256), 0, stream>>>(
        H1, W2t, b2, X1f, R2, (bf16_t*)nullptr, kBS, kD, kDFF, 0);

    // 9) LN2 -> d_out (fp32)
    ln_kernel<<<dim3(kBS), dim3(256), 0, stream>>>(R2, ln2w, ln2b, (float*)d_out,
                                                   (bf16_t*)nullptr, kD);
}
